// BackboneCausal_81020263072340
// MI455X (gfx1250) — matrix operand order repaired and checked by reading, not compiled, not run on hardware
//
#include <hip/hip_runtime.h>
#include <hip/hip_bf16.h>
#include <math.h>

// ---------------- model constants (match reference) ----------------
#define NLAY 8
#define NHEAD 16
#define HDIM 64
#define EMB 1024
#define DFF 4096
#define NFREQ 128
#define SEQ 1024
#define PAST 512
#define KTOT (PAST + SEQ)   // 1536
#define SE (SEQ * EMB)

typedef __bf16 bf16;
typedef __attribute__((ext_vector_type(16))) bf16 bf16x16;
typedef __attribute__((ext_vector_type(8))) float v8f;
typedef int v4i __attribute__((vector_size(16)));   // matches builtin param type

#define AS1 __attribute__((address_space(1)))
#define AS3 __attribute__((address_space(3)))

// CDNA5 async global->LDS path (ASYNCcnt), with synchronous fallback.
#if __has_builtin(__builtin_amdgcn_global_load_async_to_lds_b128) && \
    __has_builtin(__builtin_amdgcn_s_wait_asynccnt)
#define USE_ASYNC_LDS 1
#else
#define USE_ASYNC_LDS 0
#endif

__device__ __forceinline__ void copy16_g2l(void* lds_dst, const void* gsrc) {
#if USE_ASYNC_LDS
  __builtin_amdgcn_global_load_async_to_lds_b128(
      (AS1 v4i*)gsrc, (AS3 v4i*)lds_dst, 0, 0);
#else
  *(float4*)lds_dst = *(const float4*)gsrc;
#endif
}
__device__ __forceinline__ void wait_async() {
#if USE_ASYNC_LDS
  __builtin_amdgcn_s_wait_asynccnt(0);
#endif
}

// native f32 -> bf16 (RNE) : lowers to v_cvt_pk_bf16_f32 on gfx1250
__device__ __forceinline__ bf16 f2b(float f) { return (bf16)f; }

// =====================================================================
// 1) token embedding + fourier override
// =====================================================================
__global__ __launch_bounds__(256) void embed_fourier_kernel(
    const int* __restrict__ ids, const unsigned char* __restrict__ omask,
    const float* __restrict__ ovals, const float* __restrict__ tokemb,
    const float* __restrict__ freqs, const float* __restrict__ fpw,
    const float* __restrict__ fpb, float* __restrict__ h)
{
  const int s = blockIdx.x;
  const int tid = threadIdx.x;
  __shared__ float feats[2 * NFREQ];
  const bool ov = omask[s] != 0;
  if (ov) {
    float a = 2.0f * 3.14159265358979323846f * ovals[s] * freqs[tid & (NFREQ - 1)];
    feats[tid] = (tid < NFREQ) ? __cosf(a) : __sinf(a);
  }
  __syncthreads();
  const int id = ids[s];
  for (int e = tid; e < EMB; e += 256) {
    float v;
    if (ov) {
      float acc = fpb[e];
      const float* w = fpw + (size_t)e * (2 * NFREQ);
      #pragma unroll 8
      for (int f = 0; f < 2 * NFREQ; ++f) acc += feats[f] * w[f];
      v = acc;
    } else {
      v = tokemb[(size_t)id * EMB + e];
    }
    h[(size_t)s * EMB + e] = v;
  }
}

// =====================================================================
// 2) copy past K/V into cache layout [NL,NH,KTOT,HD] rows [0,PAST)
// =====================================================================
__global__ __launch_bounds__(256) void past_copy_kernel(
    const float* __restrict__ pk, const float* __restrict__ pv,
    float* __restrict__ Kc, float* __restrict__ Vc)
{
  size_t idx = (size_t)blockIdx.x * 256 + threadIdx.x;
  const size_t tot = (size_t)NLAY * NHEAD * PAST * HDIM;
  if (idx >= tot) return;
  int d = idx & (HDIM - 1);
  size_t r = idx >> 6;          // /HDIM
  int p = r & (PAST - 1);
  r >>= 9;                      // /PAST
  int hh = r & (NHEAD - 1);
  int l = (int)(r >> 4);
  size_t dst = (((size_t)l * NHEAD + hh) * KTOT + p) * HDIM + d;
  Kc[dst] = pk[idx];
  Vc[dst] = pv[idx];
}

// =====================================================================
// 3) RMSNorm (per-row, 256 threads, wave32 shuffle reduce)
// =====================================================================
__global__ __launch_bounds__(256) void rmsnorm_kernel(
    const float* __restrict__ x, const float* __restrict__ w,
    float* __restrict__ out)
{
  const int s = blockIdx.x, tid = threadIdx.x;
  __shared__ float red[8];
  float v[4];
  float ss = 0.f;
  #pragma unroll
  for (int q = 0; q < 4; ++q) {
    v[q] = x[(size_t)s * EMB + tid + q * 256];
    ss += v[q] * v[q];
  }
  #pragma unroll
  for (int off = 16; off > 0; off >>= 1) ss += __shfl_xor(ss, off, 32);
  if ((tid & 31) == 0) red[tid >> 5] = ss;
  __syncthreads();
  float tot = red[0] + red[1] + red[2] + red[3] + red[4] + red[5] + red[6] + red[7];
  const float inv = rsqrtf(tot * (1.0f / EMB) + 1e-6f);
  #pragma unroll
  for (int q = 0; q < 4; ++q)
    out[(size_t)s * EMB + tid + q * 256] = v[q] * inv * w[tid + q * 256];
}

// =====================================================================
// 4) bf16-WMMA GEMM:  C[M,N] = A[M,K] @ B[N,K]^T  (+ optional residual)
//    block tile 128x64, 8 waves, wave tile 32x32 (2x2 WMMA), K-chunk 32.
//    Double-buffered LDS, async global->LDS staging of raw f32,
//    f32->bf16 conversion at fragment-build time.
// =====================================================================
__global__ __launch_bounds__(256) void gemm_bf16_kernel(
    const float* __restrict__ A, const float* __restrict__ B,
    float* __restrict__ C, const float* __restrict__ R,
    int M, int N, int K, int add_res)
{
  __shared__ float As[2][128][36];   // stride 36 floats = 144B (16B aligned)
  __shared__ float Bs[2][64][36];
  const int tid  = threadIdx.x;
  const int wave = tid >> 5;
  const int lane = tid & 31;
  const int lo = lane & 15, hi = lane >> 4;
  const int m0 = blockIdx.y * 128, n0 = blockIdx.x * 64;
  const int wm = (wave >> 1) * 32, wn = (wave & 1) * 32;

  v8f acc[2][2] = {};

  auto stage = [&](int k0, int b) {
    #pragma unroll
    for (int t = 0; t < 4; ++t) {               // A: 128x8 float4
      int id = tid + t * 256;
      int row = id >> 3, c4 = (id & 7) << 2;
      copy16_g2l(&As[b][row][c4], A + (size_t)(m0 + row) * K + k0 + c4);
    }
    #pragma unroll
    for (int t = 0; t < 2; ++t) {               // B: 64x8 float4
      int id = tid + t * 256;
      int row = id >> 3, c4 = (id & 7) << 2;
      copy16_g2l(&Bs[b][row][c4], B + (size_t)(n0 + row) * K + k0 + c4);
    }
  };

  auto compute = [&](int b) {
    bf16x16 af[2], bfg[2];
    #pragma unroll
    for (int i = 0; i < 2; ++i) {               // A frag (ISA 7.12.2 layout)
      const int row = wm + i * 16 + lo;
      const int base = hi * 8;
      #pragma unroll
      for (int j = 0; j < 8; ++j) {
        af[i][j]     = f2b(As[b][row][base + j]);
        af[i][8 + j] = f2b(As[b][row][base + 16 + j]);
      }
    }
    #pragma unroll
    for (int i = 0; i < 2; ++i) {               // B frag
      const int row = wn + i * 16 + lo;
      const int base = hi * 8;
      #pragma unroll
      for (int j = 0; j < 8; ++j) {
        bfg[i][j]     = f2b(Bs[b][row][base + j]);
        bfg[i][8 + j] = f2b(Bs[b][row][base + 16 + j]);
      }
    }
    #pragma unroll
    for (int i = 0; i < 2; ++i)
      #pragma unroll
      for (int j = 0; j < 2; ++j)
        acc[i][j] = __builtin_amdgcn_wmma_f32_16x16x32_bf16(
            false, af[i], false, bfg[j], (short)0, acc[i][j], false, false);
  };

  const int nch = K >> 5;
  stage(0, 0);
  wait_async();
  __syncthreads();
  for (int c = 0; c < nch; ++c) {
    const int cur = c & 1;
    if (c + 1 < nch) stage((c + 1) << 5, cur ^ 1);  // overlap with compute
    compute(cur);
    wait_async();
    __syncthreads();
  }

  // ---- epilogue: C layout lane(lo)=N, elem r -> M = r + 8*hi ----
  #pragma unroll
  for (int i = 0; i < 2; ++i)
    #pragma unroll
    for (int j = 0; j < 2; ++j)
      #pragma unroll
      for (int r = 0; r < 8; ++r) {
        const int m = m0 + wm + i * 16 + r + 8 * hi;
        const int n = n0 + wn + j * 16 + lo;
        const size_t idx = (size_t)m * N + n;
        float v = acc[i][j][r];
        if (add_res) v += R[idx];
        C[idx] = v;
      }
}

// =====================================================================
// 5) RoPE on Q (in place) + RoPE K / copy V scattered into cache
// =====================================================================
__global__ __launch_bounds__(32) void rope_scatter_kernel(
    float* __restrict__ q, const float* __restrict__ kin,
    const float* __restrict__ vin, const float* __restrict__ rope,
    const int* __restrict__ pos, float* __restrict__ Kc, float* __restrict__ Vc)
{
  const int h = blockIdx.x, s = blockIdx.y, i = threadIdx.x; // pair index 0..31
  const int p = pos[s];
  const float c  = rope[(size_t)p * HDIM + 2 * i + 0];
  const float sn = rope[(size_t)p * HDIM + 2 * i + 1];
  const size_t qi = (size_t)s * EMB + h * HDIM + 2 * i;
  const float q0 = q[qi], q1 = q[qi + 1];
  q[qi]     = q0 * c - q1 * sn;
  q[qi + 1] = q1 * c + q0 * sn;
  const float k0 = kin[qi], k1 = kin[qi + 1];
  const size_t ko = ((size_t)h * KTOT + PAST + s) * HDIM + 2 * i;
  Kc[ko]     = k0 * c - k1 * sn;
  Kc[ko + 1] = k1 * c + k0 * sn;
  Vc[ko]     = vin[qi];
  Vc[ko + 1] = vin[qi + 1];
}

// =====================================================================
// 6) flash-style attention: 1 wave / (head, 16-query tile)
//    32-key blocks: S=Q.K^T (4 WMMA) -> online softmax -> O += P.V (4 WMMA)
//    V block staged via async global->LDS, overlapping the score WMMAs.
// =====================================================================
__global__ __launch_bounds__(32) void attn_kernel(
    const float* __restrict__ Q, const float* __restrict__ Kc,
    const float* __restrict__ Vc, float* __restrict__ AO)
{
  const int qt = blockIdx.x, head = blockIdx.y;
  const int lane = threadIdx.x;
  const int lo = lane & 15, hi = lane >> 4;
  const int qbase = qt * 16;
  const float scale = 0.125f; // 1/sqrt(64)

  __shared__ float Vs[32][68];            // 32 keys x 64 dims (+4 pad)
  __shared__ float sc[16][32];
  __shared__ bf16 pb[16][32];
  __shared__ float alpha_s[16];
  __shared__ float lsum_s[16];

  // Q fragments (A operand, rope already applied), kept live in regs
  bf16x16 aq[2];
  #pragma unroll
  for (int c = 0; c < 2; ++c) {
    const float* qrow = Q + (size_t)(qbase + lo) * EMB + head * HDIM + c * 32 + hi * 8;
    #pragma unroll
    for (int j = 0; j < 8; ++j) { aq[c][j] = f2b(qrow[j]); aq[c][8 + j] = f2b(qrow[16 + j]); }
  }

  v8f accO[4] = {};
  float m_i = -1e30f, l_i = 0.f;     // rows owned by lanes 0..15
  const int nblk = (PAST + qbase + 16 + 31) >> 5;

  for (int kb = 0; kb < nblk; ++kb) {
    const int kbase = kb * 32;

    // ---- issue async V-block stage (overlaps score WMMAs + softmax) ----
    #pragma unroll
    for (int t = 0; t < 16; ++t) {
      int idx = lane + t * 32;
      int row = idx >> 4, c4 = (idx & 15) << 2;
      copy16_g2l(&Vs[row][c4],
                 Vc + ((size_t)head * KTOT + kbase + row) * HDIM + c4);
    }

    // ---- scores: two 16-key column groups ----
    v8f s[2] = {};
    #pragma unroll
    for (int g2 = 0; g2 < 2; ++g2)
      #pragma unroll
      for (int c = 0; c < 2; ++c) {
        bf16x16 bk;
        const float* kr = Kc + ((size_t)head * KTOT + kbase + g2 * 16 + lo) * HDIM
                          + c * 32 + hi * 8;
        #pragma unroll
        for (int j = 0; j < 8; ++j) { bk[j] = f2b(kr[j]); bk[8 + j] = f2b(kr[16 + j]); }
        s[g2] = __builtin_amdgcn_wmma_f32_16x16x32_bf16(
            false, aq[c], false, bk, (short)0, s[g2], false, false);
      }
    #pragma unroll
    for (int g2 = 0; g2 < 2; ++g2)
      #pragma unroll
      for (int r = 0; r < 8; ++r) sc[r + 8 * hi][g2 * 16 + lo] = s[g2][r];
    __syncthreads();

    // ---- online softmax (lanes 0..15 each own one query row) ----
    if (lane < 16) {
      const int qabs = PAST + qbase + lane;
      float mx = m_i;
      float vv[32];
      #pragma unroll
      for (int j = 0; j < 32; ++j) {
        float x = sc[lane][j] * scale;
        if (kbase + j > qabs) x = -1e30f;
        vv[j] = x;
        if (x > mx) mx = x;
      }
      const float a = __expf(m_i - mx);
      float ls = l_i * a;
      #pragma unroll
      for (int j = 0; j < 32; ++j) {
        float p = __expf(vv[j] - mx);
        ls += p;
        pb[lane][j] = f2b(p);
      }
      m_i = mx; l_i = ls;
      alpha_s[lane] = a;
      lsum_s[lane] = ls;
    }
    wait_async();          // V block resident in LDS
    __syncthreads();

    // rescale running O (row m = r + 8*hi in C layout)
    float al[8];
    #pragma unroll
    for (int r = 0; r < 8; ++r) al[r] = alpha_s[r + 8 * hi];
    #pragma unroll
    for (int g = 0; g < 4; ++g)
      #pragma unroll
      for (int r = 0; r < 8; ++r) accO[g][r] *= al[r];

    // P as native 16x32 bf16 A-fragment
    bf16x16 ap;
    {
      const int base = hi * 8;
      #pragma unroll
      for (int j = 0; j < 8; ++j) { ap[j] = pb[lo][base + j]; ap[8 + j] = pb[lo][base + 16 + j]; }
    }
    // O += P (16x32) . V (32x64) -- 4 N-groups of 16, V frags from LDS
    #pragma unroll
    for (int g = 0; g < 4; ++g) {
      bf16x16 bv;
      #pragma unroll
      for (int j = 0; j < 8; ++j) { bv[j] = f2b(Vs[hi * 8 + j][g * 16 + lo]); bv[8 + j] = f2b(Vs[hi * 8 + 16 + j][g * 16 + lo]); }
      accO[g] = __builtin_amdgcn_wmma_f32_16x16x32_bf16(
          false, ap, false, bv, (short)0, accO[g], false, false);
    }
    __syncthreads();   // all LDS reads done before next block's async writes
  }

  // normalize and write
  #pragma unroll
  for (int g = 0; g < 4; ++g)
    #pragma unroll
    for (int r = 0; r < 8; ++r) {
      const int m = r + 8 * hi;
      const float inv = 1.0f / lsum_s[m];
      AO[(size_t)(qbase + m) * EMB + head * HDIM + g * 16 + lo] = accO[g][r] * inv;
    }
}

// =====================================================================
// 7) g = silu(g) * u
// =====================================================================
__global__ __launch_bounds__(256) void silu_mul_kernel(
    float* __restrict__ g, const float* __restrict__ u, size_t n)
{
  size_t i = (size_t)blockIdx.x * 256 + threadIdx.x;
  if (i >= n) return;
  float x = g[i];
  g[i] = (x / (1.0f + __expf(-x))) * u[i];
}

// =====================================================================
// driver
// =====================================================================
extern "C" void kernel_launch(void* const* d_in, const int* in_sizes, int n_in,
                              void* d_out, int out_size, void* d_ws, size_t ws_size,
                              hipStream_t stream) {
  (void)in_sizes; (void)n_in; (void)out_size; (void)ws_size;
  const int*   ids    = (const int*)d_in[0];
  const int*   pos    = (const int*)d_in[1];
  // d_in[2] attention_mask: causal mask recomputed analytically in-kernel
  const float* pk     = (const float*)d_in[3];
  const float* pv     = (const float*)d_in[4];
  const float* ovals  = (const float*)d_in[5];
  const unsigned char* omask = (const unsigned char*)d_in[6];
  const float* tokemb = (const float*)d_in[7];
  const float* rope   = (const float*)d_in[8];
  const float* freqs  = (const float*)d_in[9];
  const float* fpw    = (const float*)d_in[10];
  const float* fpb    = (const float*)d_in[11];
  const float* qw     = (const float*)d_in[12];
  const float* kw     = (const float*)d_in[13];
  const float* vw     = (const float*)d_in[14];
  const float* ow     = (const float*)d_in[15];
  const float* saw    = (const float*)d_in[16];
  const float* mlw    = (const float*)d_in[17];
  const float* gw     = (const float*)d_in[18];
  const float* uw     = (const float*)d_in[19];
  const float* dw     = (const float*)d_in[20];
  const float* fnw    = (const float*)d_in[21];

  float* out = (float*)d_out;
  float* h   = out;                                   // [SEQ, EMB]
  float* pks = out + (size_t)SE;                      // [NL,NH,KTOT,HD]
  float* pvs = pks + (size_t)NLAY * NHEAD * KTOT * HDIM;

  float* ws   = (float*)d_ws;                         // needs ~52 MB
  float* hn   = ws;
  float* qb   = ws + (size_t)1 * SE;
  float* kb   = ws + (size_t)2 * SE;
  float* vb   = ws + (size_t)3 * SE;
  float* ao   = ws + (size_t)4 * SE;
  float* gbuf = ws + (size_t)5 * SE;
  float* ubuf = gbuf + (size_t)SEQ * DFF;

  embed_fourier_kernel<<<SEQ, 256, 0, stream>>>(ids, omask, ovals, tokemb,
                                                freqs, fpw, fpb, h);
  {
    const size_t tot = (size_t)NLAY * NHEAD * PAST * HDIM;
    past_copy_kernel<<<(unsigned)((tot + 255) / 256), 256, 0, stream>>>(pk, pv, pks, pvs);
  }

  const dim3 gemmBlk(256);
  for (int l = 0; l < NLAY; ++l) {
    float* Kc = pks + (size_t)l * NHEAD * KTOT * HDIM;
    float* Vc = pvs + (size_t)l * NHEAD * KTOT * HDIM;
    const size_t wEE = (size_t)l * EMB * EMB;
    const size_t wFE = (size_t)l * DFF * EMB;

    rmsnorm_kernel<<<SEQ, 256, 0, stream>>>(h, saw + (size_t)l * EMB, hn);

    gemm_bf16_kernel<<<dim3(EMB / 64, SEQ / 128), gemmBlk, 0, stream>>>(
        hn, qw + wEE, qb, nullptr, SEQ, EMB, EMB, 0);
    gemm_bf16_kernel<<<dim3(EMB / 64, SEQ / 128), gemmBlk, 0, stream>>>(
        hn, kw + wEE, kb, nullptr, SEQ, EMB, EMB, 0);
    gemm_bf16_kernel<<<dim3(EMB / 64, SEQ / 128), gemmBlk, 0, stream>>>(
        hn, vw + wEE, vb, nullptr, SEQ, EMB, EMB, 0);

    rope_scatter_kernel<<<dim3(NHEAD, SEQ), 32, 0, stream>>>(
        qb, kb, vb, rope, pos, Kc, Vc);

    attn_kernel<<<dim3(SEQ / 16, NHEAD), 32, 0, stream>>>(qb, Kc, Vc, ao);

    gemm_bf16_kernel<<<dim3(EMB / 64, SEQ / 128), gemmBlk, 0, stream>>>(
        ao, ow + wEE, h, h, SEQ, EMB, EMB, 1);

    rmsnorm_kernel<<<SEQ, 256, 0, stream>>>(h, mlw + (size_t)l * EMB, hn);

    gemm_bf16_kernel<<<dim3(DFF / 64, SEQ / 128), gemmBlk, 0, stream>>>(
        hn, gw + wFE, gbuf, nullptr, SEQ, DFF, EMB, 0);
    gemm_bf16_kernel<<<dim3(DFF / 64, SEQ / 128), gemmBlk, 0, stream>>>(
        hn, uw + wFE, ubuf, nullptr, SEQ, DFF, EMB, 0);

    {
      const size_t n = (size_t)SEQ * DFF;
      silu_mul_kernel<<<(unsigned)((n + 255) / 256), 256, 0, stream>>>(gbuf, ubuf, n);
    }

    gemm_bf16_kernel<<<dim3(EMB / 64, SEQ / 128), gemmBlk, 0, stream>>>(
        gbuf, dw + (size_t)l * EMB * DFF, h, h, SEQ, EMB, DFF, 1);
  }

  rmsnorm_kernel<<<SEQ, 256, 0, stream>>>(h, fnw, h);
}
